// MoleculeNet_58093727645865
// MI455X (gfx1250) — compile-verified, matrix operand order checked by reading, not run on hardware
//
#include <hip/hip_runtime.h>
#include <hip/hip_bf16.h>
#include <math.h>

// ---------------- problem dims ----------------
#define B_  64
#define A_  127
#define E_  256
#define H_  512
#define NH_ 16
#define HD_ 32
#define L_  4
#define FF_ 2048
#define N_  128
#define M_  (B_*N_)   // 8192 rows

typedef __bf16 bf16_t;
typedef __attribute__((ext_vector_type(16))) __bf16 v16bf;
typedef __attribute__((ext_vector_type(8)))  __bf16 v8bf;
typedef __attribute__((ext_vector_type(8)))  float  v8f;

static __device__ inline float  bf2f(bf16_t v){ return (float)v; }
static __device__ inline bf16_t f2bf(float v){ return (bf16_t)v; }

// ---------------- WMMA helpers ----------------
// 16-bit A/B fragment, 16x32 tile, row-major source with leading dim ld.
// Per ISA 7.12.2: lane half 0 covers K = {k0..k0+7, k0+16..k0+23},
// lane half 1 covers K = {k0+8..k0+15, k0+24..k0+31}; row = row0 + (lane&15).
static __device__ inline v16bf frag_ab(const bf16_t* __restrict__ base, int row0, int ld, int k0)
{
    int lane = threadIdx.x & 31;
    int r    = row0 + (lane & 15);
    int half = lane >> 4;
    const bf16_t* p = base + (size_t)r * ld + k0 + half * 8;
    v8bf lo = *(const v8bf*)p;
    v8bf hi = *(const v8bf*)(p + 16);
    v16bf f;
#pragma unroll
    for (int i = 0; i < 8; ++i) { f[i] = lo[i]; f[i + 8] = hi[i]; }
    return f;
}

static __device__ inline v8f wmma_bf16(v16bf a, v16bf b, v8f c)
{
    return __builtin_amdgcn_wmma_f32_16x16x32_bf16(false, a, false, b, (short)0, c, false, false);
}

static __device__ inline v8f zero8()
{
    v8f c;
#pragma unroll
    for (int i = 0; i < 8; ++i) c[i] = 0.f;
    return c;
}

// ---------------- wave reduce ----------------
static __device__ inline float wsum(float v){
#pragma unroll
    for (int m = 16; m > 0; m >>= 1) v += __shfl_xor(v, m, 32);
    return v;
}
static __device__ inline float wmaxf(float v){
#pragma unroll
    for (int m = 16; m > 0; m >>= 1) v = fmaxf(v, __shfl_xor(v, m, 32));
    return v;
}

// ---------------- block LN over 512 (256 threads, 2 elems each) ----------------
static __device__ inline void block_ln512(float& v0, float& v1, const float* g, const float* bvec,
                                          float eps, float* red, float* red2)
{
    int tid = threadIdx.x;
    red[tid]  = v0 + v1;
    red2[tid] = v0 * v0 + v1 * v1;
    __syncthreads();
#pragma unroll
    for (int ofs = 128; ofs > 0; ofs >>= 1) {
        if (tid < ofs) { red[tid] += red[tid + ofs]; red2[tid] += red2[tid + ofs]; }
        __syncthreads();
    }
    float mean = red[0] * (1.0f / 512.0f);
    float var  = red2[0] * (1.0f / 512.0f) - mean * mean;
    float rstd = rsqrtf(var + eps);
    v0 = (v0 - mean) * rstd * g[tid]       + bvec[tid];
    v1 = (v1 - mean) * rstd * g[tid + 256] + bvec[tid + 256];
    __syncthreads();
}

// ================= kernels =================

// atom embedding -> hidden f32 (d_out) + bf16 copy. One block per (b,node).
__global__ void embed_kernel(const int* __restrict__ acate, const float* __restrict__ afloat,
                             const float* __restrict__ ctab, const float* __restrict__ clg,
                             const float* __restrict__ clb, const float* __restrict__ fW,
                             const float* __restrict__ fb, const float* __restrict__ flg,
                             const float* __restrict__ flb,
                             float* __restrict__ hidden, bf16_t* __restrict__ xb)
{
    __shared__ float red[256], red2[256];
    int row = blockIdx.x;            // b*N + n
    int b = row >> 7, n = row & 127;
    int tid = threadIdx.x;
    size_t base = (size_t)row * H_;
    if (n == 0) {
        hidden[base + tid] = 0.f; hidden[base + tid + 256] = 0.f;
        xb[base + tid] = f2bf(0.f); xb[base + tid + 256] = f2bf(0.f);
        return;
    }
    int a = n - 1;
    float s0 = 0.f, s1 = 0.f;
#pragma unroll
    for (int f = 0; f < 9; ++f) {
        int idx = acate[((size_t)b * A_ + a) * 9 + f] + 16 * f;
        s0 += ctab[(size_t)idx * H_ + tid];
        s1 += ctab[(size_t)idx * H_ + tid + 256];
    }
    block_ln512(s0, s1, clg, clb, 1e-5f, red, red2);
    float t0 = fb[tid], t1 = fb[tid + 256];
#pragma unroll
    for (int i = 0; i < 16; ++i) {
        float av = afloat[((size_t)b * A_ + a) * 16 + i];
        t0 += av * fW[(size_t)i * H_ + tid];
        t1 += av * fW[(size_t)i * H_ + tid + 256];
    }
    block_ln512(t0, t1, flg, flb, 1e-5f, red, red2);
    float h0 = s0 + t0, h1 = s1 + t1;
    hidden[base + tid] = h0; hidden[base + tid + 256] = h1;
    xb[base + tid] = f2bf(h0); xb[base + tid + 256] = f2bf(h1);
}

// residual + LN (H=512): y = LN(x+res); writes f32 and bf16.
__global__ void ln_res_kernel(const float* __restrict__ x, const float* __restrict__ res,
                              const float* __restrict__ g, const float* __restrict__ bb,
                              float eps, float* __restrict__ outf, bf16_t* __restrict__ outb)
{
    __shared__ float red[256], red2[256];
    int row = blockIdx.x, tid = threadIdx.x;
    size_t base = (size_t)row * H_;
    float v0 = x[base + tid] + res[base + tid];
    float v1 = x[base + tid + 256] + res[base + tid + 256];
    block_ln512(v0, v1, g, bb, eps, red, red2);
    outf[base + tid] = v0; outf[base + tid + 256] = v1;
    outb[base + tid] = f2bf(v0); outb[base + tid + 256] = f2bf(v1);
}

// transpose + f32->bf16 convert: W[K,N] -> Wt[N,K]
__global__ void convT_kernel(const float* __restrict__ W, bf16_t* __restrict__ Wt, int K, int N)
{
    int i = blockIdx.x * 256 + threadIdx.x;
    if (i >= K * N) return;
    int n = i % N, k = i / N;
    Wt[(size_t)n * K + k] = f2bf(W[i]);
}

// bond embedding hb[b,e,HD]: one wave per (b,e), lane = dim; wave LN over 32.
__global__ void bondemb_kernel(const int* __restrict__ bcate, const float* __restrict__ bfl,
                               const float* __restrict__ btab, const float* __restrict__ clg,
                               const float* __restrict__ clb, const float* __restrict__ fW,
                               const float* __restrict__ fb, const float* __restrict__ flg,
                               const float* __restrict__ flb, float* __restrict__ hb)
{
    int gw = blockIdx.x * 8 + (threadIdx.x >> 5);
    if (gw >= B_ * E_) return;
    int d = threadIdx.x & 31;
    float s = 0.f;
#pragma unroll
    for (int f = 0; f < 3; ++f) {
        int idx = bcate[(size_t)gw * 3 + f] + 8 * f;
        s += btab[(size_t)idx * HD_ + d];
    }
    float m  = wsum(s) * (1.f / 32.f);
    float v  = wsum(s * s) * (1.f / 32.f) - m * m;
    float r1 = (s - m) * rsqrtf(v + 1e-5f) * clg[d] + clb[d];
    float t = fb[d];
#pragma unroll
    for (int i = 0; i < 8; ++i) t += bfl[(size_t)gw * 8 + i] * fW[(size_t)i * HD_ + d];
    float m2 = wsum(t) * (1.f / 32.f);
    float v2 = wsum(t * t) * (1.f / 32.f) - m2 * m2;
    float r2 = (t - m2) * rsqrtf(v2 + 1e-5f) * flg[d] + flb[d];
    hb[(size_t)gw * HD_ + d] = r1 + r2;
}

// vals[b,e,:] = hb[b, bond_idx[b,e], :]
__global__ void vals_kernel(const float* __restrict__ hb, const int* __restrict__ bidx,
                            float* __restrict__ vals)
{
    int i = blockIdx.x * 256 + threadIdx.x;
    if (i >= B_ * E_ * HD_) return;
    int d = i & 31, be = i >> 5;
    int b = be / E_;
    int src = bidx[be];
    vals[i] = hb[((size_t)b * E_ + src) * HD_ + d];
}

// LN the small path(32x32) and ring(8x32) tables; wave per row.
__global__ void lnpr_kernel(const float* __restrict__ ptab, const float* __restrict__ plg,
                            const float* __restrict__ plb, const float* __restrict__ rtab,
                            const float* __restrict__ rlg, const float* __restrict__ rlb,
                            float* __restrict__ lnpath, float* __restrict__ lnring)
{
    int w = threadIdx.x >> 5, d = threadIdx.x & 31;
    for (int row = w; row < 40; row += 8) {
        const float *src, *g, *bb; float* dst;
        if (row < 32) { src = ptab + row * 32; g = plg; bb = plb; dst = lnpath + row * 32; }
        else          { src = rtab + (row - 32) * 32; g = rlg; bb = rlb; dst = lnring + (row - 32) * 32; }
        float v = src[d];
        float m  = wsum(v) * (1.f / 32.f);
        float vr = wsum(v * v) * (1.f / 32.f) - m * m;
        dst[d] = (v - m) * rsqrtf(vr + 1e-5f) * g[d] + bb[d];
    }
}

// Tiled WMMA GEMM: C[M,N] = A[M,K](bf16) @ Bt[N,K](bf16)^T + bias.
// Each wave computes a 32x64 macro-tile (2 m-tiles x 4 n-tiles, 8 accumulators):
// 8 WMMAs per K-step from 6 fragment loads (vs 1 WMMA / 2 loads before).
template<int ACT, int OUTBF>
__global__ void gemm_kernel(const bf16_t* __restrict__ A, const bf16_t* __restrict__ Bt,
                            const float* __restrict__ bias, void* __restrict__ Cout,
                            int M, int N, int K)
{
    int wave = blockIdx.x * 8 + (threadIdx.x >> 5);
    int nMac = N >> 6;                      // 64-col macro tiles along N
    int tM = wave / nMac, tN = wave % nMac; // tM: 32-row macro index
    if (tM >= (M >> 5)) return;
    int lane = threadIdx.x & 31;
    int half = lane >> 4, nl = lane & 15;
    v8f acc[2][4];
#pragma unroll
    for (int i = 0; i < 2; ++i)
#pragma unroll
        for (int j = 0; j < 4; ++j) acc[i][j] = zero8();
    int rowA0 = tM * 32, rowB0 = tN * 64;
    for (int k0 = 0; k0 < K; k0 += 32) {
        __builtin_prefetch(A  + (size_t)(rowA0 + nl) * K + k0 + 32, 0, 1);
        __builtin_prefetch(Bt + (size_t)(rowB0 + nl) * K + k0 + 32, 0, 1);
        v16bf a0 = frag_ab(A, rowA0,      K, k0);
        v16bf a1 = frag_ab(A, rowA0 + 16, K, k0);
#pragma unroll
        for (int j = 0; j < 4; ++j) {
            v16bf b = frag_ab(Bt, rowB0 + j * 16, K, k0);
            acc[0][j] = wmma_bf16(a0, b, acc[0][j]);
            acc[1][j] = wmma_bf16(a1, b, acc[1][j]);
        }
    }
#pragma unroll
    for (int i = 0; i < 2; ++i) {
#pragma unroll
        for (int j = 0; j < 4; ++j) {
            int col = rowB0 + j * 16 + nl;
            float bv = bias ? bias[col] : 0.f;
#pragma unroll
            for (int r = 0; r < 8; ++r) {
                int row = rowA0 + i * 16 + r + half * 8;
                float v = acc[i][j][r] + bv;
                if (ACT == 1) v = 0.5f * v * (1.f + erff(v * 0.70710678118654752f)); // exact GELU
                if (OUTBF) ((bf16_t*)Cout)[(size_t)row * N + col] = f2bf(v);
                else       ((float*)Cout)[(size_t)row * N + col]  = v;
            }
        }
    }
}

// Fused attention per (b,h): S=QK^T (WMMA) + pe bias via dp/dr WMMAs + sparse bond dots,
// softmax, ctx = P@V (WMMA). Q/K tiles staged to LDS with async global->LDS copies.
__global__ void attn_kernel(const bf16_t* __restrict__ qb, const bf16_t* __restrict__ kb,
                            const bf16_t* __restrict__ vb, const float* __restrict__ amask,
                            const int* __restrict__ spl, const int* __restrict__ ring,
                            const int* __restrict__ eidx, const float* __restrict__ vals,
                            const float* __restrict__ lnpath, const float* __restrict__ lnring,
                            bf16_t* __restrict__ ctxb)
{
    extern __shared__ char smem[];
    float*  sS   = (float*)smem;                 // 128*128 f32
    float*  sDP  = sS  + 128 * 128;              // 128*32
    float*  sDR  = sDP + 128 * 32;               // 128*16
    bf16_t* sP   = (bf16_t*)(sDR + 128 * 16);    // 128*128 bf16
    bf16_t* sQ   = sP  + 128 * 128;              // 128*32
    bf16_t* sK   = sQ  + 128 * 32;               // 128*32
    bf16_t* sVT  = sK  + 128 * 32;               // 32*128 (V transposed)
    bf16_t* sLP  = sVT + 32 * 128;               // 32*32
    bf16_t* sLR  = sLP + 32 * 32;                // 16*32 (8 real + 8 pad)
    float*  smask= (float*)(sLR + 16 * 32);      // 128

    int bh = blockIdx.x;
    int b = bh / NH_, h = bh % NH_;
    int tid = threadIdx.x;

    // Q,K: async global->LDS (16B per lane per issue, ASYNCcnt-tracked)
    for (int idx = tid; idx < 128 * 4; idx += 256) {
        int n = idx >> 2, ch = idx & 3;
        size_t go = ((size_t)(b * N_ + n)) * H_ + h * HD_ + ch * 8;
        unsigned ldsq = (unsigned)((char*)(sQ + n * 32 + ch * 8) - smem);
        unsigned ldsk = (unsigned)((char*)(sK + n * 32 + ch * 8) - smem);
        unsigned long long gq = (unsigned long long)(qb + go);
        unsigned long long gk = (unsigned long long)(kb + go);
        asm volatile("global_load_async_to_lds_b128 %0, %1, off"
                     :: "v"(ldsq), "v"(gq) : "memory");
        asm volatile("global_load_async_to_lds_b128 %0, %1, off"
                     :: "v"(ldsk), "v"(gk) : "memory");
    }
    for (int idx = tid; idx < 128 * 32; idx += 256) {    // V transposed
        int kk = idx >> 5, d = idx & 31;
        sVT[d * 128 + kk] = vb[((size_t)(b * N_ + kk)) * H_ + h * HD_ + d];
    }
    for (int idx = tid; idx < 32 * 32; idx += 256) sLP[idx] = f2bf(lnpath[idx]);
    for (int idx = tid; idx < 16 * 32; idx += 256) sLR[idx] = (idx < 8 * 32) ? f2bf(lnring[idx]) : f2bf(0.f);
    for (int k = tid; k < 128; k += 256)
        smask[k] = (k == 0) ? 0.f : (amask[(size_t)b * A_ + k - 1] - 1.f) * 1e9f;
    asm volatile("s_wait_asynccnt 0" ::: "memory");
    __syncthreads();

    int w = tid >> 5;
    int lane = tid & 31, half = lane >> 4, nl = lane & 15;

    {   // S = Q K^T ; dp = Q @ lnpath^T ; dr = Q @ lnring^T   (wave w owns q-tile w)
        int qt = w;
        v16bf aq = frag_ab(sQ, qt * 16, 32, 0);
        for (int kt = 0; kt < 8; ++kt) {
            v16bf bk2 = frag_ab(sK, kt * 16, 32, 0);
            v8f c = zero8();
            c = wmma_bf16(aq, bk2, c);
#pragma unroll
            for (int r = 0; r < 8; ++r)
                sS[(qt * 16 + r + half * 8) * 128 + kt * 16 + nl] = c[r];
        }
        for (int nt = 0; nt < 2; ++nt) {
            v16bf bp = frag_ab(sLP, nt * 16, 32, 0);
            v8f c = zero8();
            c = wmma_bf16(aq, bp, c);
#pragma unroll
            for (int r = 0; r < 8; ++r)
                sDP[(qt * 16 + r + half * 8) * 32 + nt * 16 + nl] = c[r];
        }
        {
            v16bf br = frag_ab(sLR, 0, 32, 0);
            v8f c = zero8();
            c = wmma_bf16(aq, br, c);
#pragma unroll
            for (int r = 0; r < 8; ++r)
                sDR[(qt * 16 + r + half * 8) * 16 + nl] = c[r];
        }
    }
    __syncthreads();

    // pe bias: S[q,k] += dp[q, spl[q-1,k-1]] + dr[q, ring[q-1,k-1]]  for q,k >= 1
    for (int idx = tid; idx < 128 * 128; idx += 256) {
        int q = idx >> 7, k = idx & 127;
        if (q >= 1 && k >= 1) {
            size_t o = ((size_t)b * A_ + (q - 1)) * A_ + (k - 1);
            sS[idx] += sDP[q * 32 + spl[o]] + sDR[q * 16 + ring[o]];
        }
    }
    __syncthreads();

    // sparse bond-embedding bias
    for (int e = tid; e < E_; e += 256) {
        int r = eidx[(size_t)b * 2 * E_ + e] + 1;
        int c = eidx[(size_t)b * 2 * E_ + E_ + e] + 1;
        const bf16_t* qrow = qb + ((size_t)(b * N_ + r)) * H_ + h * HD_;
        const float*  vv   = vals + ((size_t)b * E_ + e) * HD_;
        float acc = 0.f;
#pragma unroll
        for (int d = 0; d < HD_; ++d) acc += bf2f(qrow[d]) * vv[d];
        sS[r * 128 + c] += acc;
    }
    __syncthreads();

    // softmax: wave w handles rows [w*16, w*16+16); lane covers 4 cols
    const float scale = 0.17677669529663687f;  // 1/sqrt(32)
    for (int i = 0; i < 16; ++i) {
        int q = w * 16 + i;
        float sv[4], mx = -1e30f;
#pragma unroll
        for (int j = 0; j < 4; ++j) {
            int c = lane + j * 32;
            sv[j] = sS[q * 128 + c] * scale + smask[c];
            mx = fmaxf(mx, sv[j]);
        }
        mx = wmaxf(mx);
        float se = 0.f;
#pragma unroll
        for (int j = 0; j < 4; ++j) { sv[j] = __expf(sv[j] - mx); se += sv[j]; }
        se = wsum(se);
        float inv = 1.f / se;
#pragma unroll
        for (int j = 0; j < 4; ++j) sP[q * 128 + lane + j * 32] = f2bf(sv[j] * inv);
    }
    __syncthreads();

    {   // ctx = P @ V
        int qt = w;
        for (int nt = 0; nt < 2; ++nt) {
            v8f c = zero8();
            for (int kb2 = 0; kb2 < 4; ++kb2) {
                v16bf a  = frag_ab(sP,  qt * 16, 128, kb2 * 32);
                v16bf bb = frag_ab(sVT, nt * 16, 128, kb2 * 32);
                c = wmma_bf16(a, bb, c);
            }
#pragma unroll
            for (int r = 0; r < 8; ++r) {
                int q   = qt * 16 + r + half * 8;
                int col = nt * 16 + nl;
                ctxb[((size_t)(b * N_ + q)) * H_ + h * HD_ + col] = f2bf(c[r]);
            }
        }
    }
}

// ================= host =================
extern "C" void kernel_launch(void* const* d_in, const int* in_sizes, int n_in,
                              void* d_out, int out_size, void* d_ws, size_t ws_size,
                              hipStream_t stream)
{
    (void)in_sizes; (void)n_in; (void)out_size; (void)ws_size;
    enum { I_ACATE=0, I_AFLOAT, I_AMASK, I_BCATE, I_BFLOAT, I_BMASK, I_EIAB, I_EMAB,
           I_EIAA, I_EMAA, I_EBOND, I_SPL, I_RING, I_PATHS2, I_GCATE, I_GFP,
           P_ACATE_TAB, P_ACATE_LG, P_ACATE_LB, P_AF_W, P_AF_B, P_AF_LG, P_AF_LB,
           P_PATH_TAB, P_PATH_LG, P_PATH_LB, P_RING_TAB, P_RING_LG, P_RING_LB,
           P_BOND_TAB, P_BC_LG, P_BC_LB, P_BF_W, P_BF_B, P_BF_LG, P_BF_LB,
           P_WQ, P_BQ, P_WK, P_BK, P_WV, P_BV, P_WO, P_BO, P_LN1G, P_LN1B,
           P_W1, P_B1, P_W2, P_B2, P_LN2G, P_LN2B };

    const int*   acate  = (const int*)  d_in[I_ACATE];
    const float* afloat = (const float*)d_in[I_AFLOAT];
    const float* amask  = (const float*)d_in[I_AMASK];
    const int*   bcate  = (const int*)  d_in[I_BCATE];
    const float* bfl    = (const float*)d_in[I_BFLOAT];
    const int*   eiaa   = (const int*)  d_in[I_EIAA];
    const int*   ebond  = (const int*)  d_in[I_EBOND];
    const int*   spl    = (const int*)  d_in[I_SPL];
    const int*   ringc  = (const int*)  d_in[I_RING];

    float* hidden = (float*)d_out;   // [B,N,H] fp32 master

    // workspace bump allocator
    char* ws = (char*)d_ws;
    auto alloc = [&](size_t bytes) -> void* {
        void* p = (void*)ws; ws += (bytes + 255) & ~(size_t)255; return p;
    };
    bf16_t* xb    = (bf16_t*)alloc((size_t)M_ * H_ * 2);
    bf16_t* qbuf  = (bf16_t*)alloc((size_t)M_ * H_ * 2);
    bf16_t* kbuf  = (bf16_t*)alloc((size_t)M_ * H_ * 2);
    bf16_t* vbuf  = (bf16_t*)alloc((size_t)M_ * H_ * 2);
    bf16_t* ctxb  = (bf16_t*)alloc((size_t)M_ * H_ * 2);
    bf16_t* attnb = (bf16_t*)alloc((size_t)M_ * H_ * 2);
    bf16_t* ff1b  = (bf16_t*)alloc((size_t)M_ * FF_ * 2);
    float*  tmpf  = (float*) alloc((size_t)M_ * H_ * 4);
    float*  attnf = (float*) alloc((size_t)M_ * H_ * 4);
    float*  hbbuf = (float*) alloc((size_t)B_ * E_ * HD_ * 4);
    float*  valsb = (float*) alloc((size_t)B_ * E_ * HD_ * 4);
    float*  lnpathb = (float*)alloc(32 * 32 * 4);
    float*  lnringb = (float*)alloc(8 * 32 * 4);
    bf16_t* wqb = (bf16_t*)alloc((size_t)H_ * H_ * 2);
    bf16_t* wkb = (bf16_t*)alloc((size_t)H_ * H_ * 2);
    bf16_t* wvb = (bf16_t*)alloc((size_t)H_ * H_ * 2);
    bf16_t* wob = (bf16_t*)alloc((size_t)H_ * H_ * 2);
    bf16_t* w1b = (bf16_t*)alloc((size_t)H_ * FF_ * 2);
    bf16_t* w2b = (bf16_t*)alloc((size_t)FF_ * H_ * 2);

    const size_t ATTN_SMEM =
        128*128*4 + 128*32*4 + 128*16*4 +      // sS, sDP, sDR
        (size_t)128*128*2 + 128*32*2*2 +       // sP, sQ, sK
        32*128*2 + 32*32*2 + 16*32*2 + 128*4;  // sVT, sLP, sLR, smask

    // grid sizes for 32x64-per-wave GEMM, 8 waves per block
    const int GRID_H  = (M_/32) * (H_/64)  / 8;   // N=512  -> 256 blocks
    const int GRID_FF = (M_/32) * (FF_/64) / 8;   // N=2048 -> 1024 blocks

    // atom embedding
    embed_kernel<<<M_, 256, 0, stream>>>(acate, afloat,
        (const float*)d_in[P_ACATE_TAB], (const float*)d_in[P_ACATE_LG], (const float*)d_in[P_ACATE_LB],
        (const float*)d_in[P_AF_W], (const float*)d_in[P_AF_B],
        (const float*)d_in[P_AF_LG], (const float*)d_in[P_AF_LB], hidden, xb);

    for (int l = 0; l < L_; ++l) {
        const float* Wq = (const float*)d_in[P_WQ] + (size_t)l * H_ * H_;
        const float* Wk = (const float*)d_in[P_WK] + (size_t)l * H_ * H_;
        const float* Wv = (const float*)d_in[P_WV] + (size_t)l * H_ * H_;
        const float* Wo = (const float*)d_in[P_WO] + (size_t)l * H_ * H_;
        const float* W1 = (const float*)d_in[P_W1] + (size_t)l * H_ * FF_;
        const float* W2 = (const float*)d_in[P_W2] + (size_t)l * FF_ * H_;
        const float* bq = (const float*)d_in[P_BQ] + (size_t)l * H_;
        const float* bk = (const float*)d_in[P_BK] + (size_t)l * H_;
        const float* bv = (const float*)d_in[P_BV] + (size_t)l * H_;
        const float* bo = (const float*)d_in[P_BO] + (size_t)l * H_;
        const float* b1 = (const float*)d_in[P_B1] + (size_t)l * FF_;
        const float* b2 = (const float*)d_in[P_B2] + (size_t)l * H_;

        convT_kernel<<<(H_*H_)/256, 256, 0, stream>>>(Wq, wqb, H_, H_);
        convT_kernel<<<(H_*H_)/256, 256, 0, stream>>>(Wk, wkb, H_, H_);
        convT_kernel<<<(H_*H_)/256, 256, 0, stream>>>(Wv, wvb, H_, H_);
        convT_kernel<<<(H_*H_)/256, 256, 0, stream>>>(Wo, wob, H_, H_);
        convT_kernel<<<(H_*FF_)/256, 256, 0, stream>>>(W1, w1b, H_, FF_);
        convT_kernel<<<(FF_*H_)/256, 256, 0, stream>>>(W2, w2b, FF_, H_);

        bondemb_kernel<<<(B_*E_)/8, 256, 0, stream>>>(bcate, bfl,
            (const float*)d_in[P_BOND_TAB] + (size_t)l * 24 * HD_,
            (const float*)d_in[P_BC_LG] + (size_t)l * HD_, (const float*)d_in[P_BC_LB] + (size_t)l * HD_,
            (const float*)d_in[P_BF_W] + (size_t)l * 8 * HD_, (const float*)d_in[P_BF_B] + (size_t)l * HD_,
            (const float*)d_in[P_BF_LG] + (size_t)l * HD_, (const float*)d_in[P_BF_LB] + (size_t)l * HD_,
            hbbuf);
        vals_kernel<<<(B_*E_*HD_)/256, 256, 0, stream>>>(hbbuf, ebond, valsb);
        lnpr_kernel<<<1, 256, 0, stream>>>(
            (const float*)d_in[P_PATH_TAB] + (size_t)l * 32 * HD_,
            (const float*)d_in[P_PATH_LG] + (size_t)l * HD_, (const float*)d_in[P_PATH_LB] + (size_t)l * HD_,
            (const float*)d_in[P_RING_TAB] + (size_t)l * 8 * HD_,
            (const float*)d_in[P_RING_LG] + (size_t)l * HD_, (const float*)d_in[P_RING_LB] + (size_t)l * HD_,
            lnpathb, lnringb);

        // QKV projections (bf16 out)
        gemm_kernel<0,1><<<GRID_H, 256, 0, stream>>>(xb, wqb, bq, qbuf, M_, H_, H_);
        gemm_kernel<0,1><<<GRID_H, 256, 0, stream>>>(xb, wkb, bk, kbuf, M_, H_, H_);
        gemm_kernel<0,1><<<GRID_H, 256, 0, stream>>>(xb, wvb, bv, vbuf, M_, H_, H_);

        attn_kernel<<<B_*NH_, 256, ATTN_SMEM, stream>>>(qbuf, kbuf, vbuf, amask,
            spl, ringc, eiaa, valsb, lnpathb, lnringb, ctxb);

        // output projection + residual LN
        gemm_kernel<0,0><<<GRID_H, 256, 0, stream>>>(ctxb, wob, bo, tmpf, M_, H_, H_);
        ln_res_kernel<<<M_, 256, 0, stream>>>(tmpf, hidden, (const float*)d_in[P_LN1G] + (size_t)l * H_,
            (const float*)d_in[P_LN1B] + (size_t)l * H_, 1e-12f, attnf, attnb);

        // FFN
        gemm_kernel<1,1><<<GRID_FF, 256, 0, stream>>>(attnb, w1b, b1, ff1b, M_, FF_, H_);
        gemm_kernel<0,0><<<GRID_H, 256, 0, stream>>>(ff1b, w2b, b2, tmpf, M_, H_, FF_);
        ln_res_kernel<<<M_, 256, 0, stream>>>(tmpf, attnf, (const float*)d_in[P_LN2G] + (size_t)l * H_,
            (const float*)d_in[P_LN2B] + (size_t)l * H_, 1e-12f, hidden, xb);
    }
}
